// Attention_27702539059755
// MI455X (gfx1250) — matrix operand order repaired and checked by reading, not compiled, not run on hardware
//
#include <hip/hip_runtime.h>
#include <hip/hip_bf16.h>
#include <math.h>

typedef _Float16 f16;
typedef __attribute__((ext_vector_type(8)))  _Float16 v8h_t;
typedef __attribute__((ext_vector_type(16))) _Float16 v16h_t;
typedef __attribute__((ext_vector_type(8)))  float    v8f_t;

#define BB   128
#define NN   64
#define EE   128
#define NE   63           // N-1 edges per agent
#define MROW 8192         // B*N
#define MEDG 516096       // B*N*(N-1)

// d_out layout (flat, f32)
#define OUT_ATT 0
#define OUT_HL  2097152
#define OUT_URD 2613248
#define OUT_ENT 3129344
#define OUT_HW  3129345
#define OUT_CW  3645441

__device__ __forceinline__ float lrelu(float x) { return x >= 0.f ? x : 0.01f * x; }

// ---------------------------------------------------------------------------
// Weight prep: f32 -> f16 copies, split Wh1/Wa1 into [ae-part | edge-part],
// fold Whe@Wh2 into Wcomb (2x128), bcomb = Whe@bh2 + bhe.
// ---------------------------------------------------------------------------
__global__ void prep_weights(const float* W2, const float* Wq, const float* Wh1,
                             const float* Wa1, const float* Wh2, const float* bh2,
                             const float* Whe, const float* bhe,
                             const float* Wd1, const float* Wd2,
                             f16* W2h, f16* Wqh, f16* Wh1ah, float* Wh1e,
                             f16* Wa1ah, float* Wa1e, f16* Wd1h, f16* Wd2h,
                             float* Wcomb, float* bcomb) {
  int gtid = blockIdx.x * blockDim.x + threadIdx.x;
  int T = gridDim.x * blockDim.x;
  for (int i = gtid; i < 128 * 128; i += T) {
    W2h[i] = (f16)W2[i];
    Wqh[i] = (f16)Wq[i];
    int n = i >> 7, k = i & 127;
    Wh1ah[i] = (f16)Wh1[n * 135 + k];
    Wa1ah[i] = (f16)Wa1[n * 135 + k];
  }
  for (int i = gtid; i < 128 * 7; i += T) {
    int n = i / 7, f = i % 7;
    Wh1e[i] = Wh1[n * 135 + 128 + f];
    Wa1e[i] = Wa1[n * 135 + 128 + f];
  }
  for (int i = gtid; i < 256 * 256; i += T) {
    Wd1h[i] = (f16)Wd1[i];
    Wd2h[i] = (f16)Wd2[i];
  }
  for (int i = gtid; i < 2 * 128; i += T) {
    int r = i >> 7, t = i & 127;
    float s = 0.f;
    for (int u = 0; u < 128; ++u) s += Whe[r * 128 + u] * Wh2[u * 128 + t];
    Wcomb[i] = s;
  }
  if (gtid < 2) {
    float s = bhe[gtid];
    for (int u = 0; u < 128; ++u) s += Whe[gtid * 128 + u] * bh2[u];
    bcomb[gtid] = s;
  }
}

// ---------------------------------------------------------------------------
// Layer 1 of agent MLP: h1 = lrelu(embed(5) @ W1^T + b1), stored f16.
// ---------------------------------------------------------------------------
__global__ void h1_kernel(const float* emb, const float* W1, const float* b1, f16* h1h) {
  int gtid = blockIdx.x * blockDim.x + threadIdx.x;
  int T = gridDim.x * blockDim.x;
  for (int i = gtid; i < MROW * 128; i += T) {
    int m = i >> 7, t = i & 127;
    const float* e = emb + m * 9 + 4;
    const float* w = W1 + t * 5;
    float h = b1[t] + e[0] * w[0] + e[1] * w[1] + e[2] * w[2] + e[3] * w[3] + e[4] * w[4];
    h1h[i] = (f16)lrelu(h);
  }
}

// ---------------------------------------------------------------------------
// WMMA GEMM: C(M,N) = act(A(M,K) @ W(N,K)^T + bias). A,W f16; C f32 and/or f16.
// One 16x16 tile per wave; 4 waves per block. M,N multiples of 16; K mult of 32.
// act: 0 = none, 1 = leaky_relu(0.01)
// ---------------------------------------------------------------------------
__global__ __launch_bounds__(128) void wmma_gemm(const f16* __restrict__ A,
                                                 const f16* __restrict__ W,
                                                 const float* __restrict__ bias,
                                                 float* __restrict__ outF,
                                                 f16* __restrict__ outH,
                                                 int K, int N, int ntn, int act) {
  int wid = threadIdx.x >> 5;
  int tile = blockIdx.x * 4 + wid;
  int tm = tile / ntn;
  int tn = tile % ntn;
  int lane = threadIdx.x & 31;
  int half = lane >> 4;
  int m = tm * 16 + (lane & 15);
  int n = tn * 16 + (lane & 15);

  v8f_t c = {};
  int kc_count = K >> 5;
  const f16* arow = A + (size_t)m * K;
  const f16* wrow = W + (size_t)n * K;
  for (int kc = 0; kc < kc_count; ++kc) {
    int kb = kc * 32;
    v8h_t a0 = *(const v8h_t*)(arow + kb + half * 8);
    v8h_t a1 = *(const v8h_t*)(arow + kb + 16 + half * 8);
    v16h_t av = __builtin_shufflevector(a0, a1, 0, 1, 2, 3, 4, 5, 6, 7, 8, 9, 10, 11, 12, 13, 14, 15);
    v8h_t b0 = *(const v8h_t*)(wrow + kb + half * 8);
    v8h_t b1 = *(const v8h_t*)(wrow + kb + 16 + half * 8);
    v16h_t bvv = __builtin_shufflevector(b0, b1, 0, 1, 2, 3, 4, 5, 6, 7, 8, 9, 10, 11, 12, 13, 14, 15);
    c = __builtin_amdgcn_wmma_f32_16x16x32_f16(false, av, false, bvv, (short)0, c, false, false);
  }

  float bs = bias ? bias[n] : 0.f;
#pragma unroll
  for (int r = 0; r < 8; ++r) {
    float v = c[r] + bs;
    if (act == 1) v = lrelu(v);
    size_t row = (size_t)(tm * 16 + half * 8 + r);
    if (outF) outF[row * N + n] = v;
    if (outH) outH[row * N + n] = (f16)v;
  }
}

// ---------------------------------------------------------------------------
// Fused per-agent edge kernel: one block (64 threads) per (b,i).
// Thread j < 63 owns edge j. Computes edge features, gate logits, scores,
// masked softmax, entropy, weighted edge sums -> attn_out -> concat buffer.
// ---------------------------------------------------------------------------
__global__ __launch_bounds__(64) void edge_kernel(
    const float* __restrict__ emb, const float* __restrict__ gum,
    const float* __restrict__ Ph, const float* __restrict__ Pa,
    const float* __restrict__ Wh1e, const float* __restrict__ Wa1e,
    const float* __restrict__ Wcomb, const float* __restrict__ bcomb,
    const float* __restrict__ Wa2, const float* __restrict__ ba2,
    const float* __restrict__ Wv, const float* __restrict__ bv,
    const f16* __restrict__ ae_h, f16* __restrict__ concat_h,
    float* __restrict__ out, float* __restrict__ ent_ws) {
  __shared__ float sWh1e[896], sWa1e[896], sWv[896];
  __shared__ float sWcomb[256], sWa2[128], sBv[128], sPh[128], sPa[128];
  __shared__ float sMS[64], sVal[64], sMask[64];
  __shared__ float sEdge[NE * 7];
  __shared__ float sBroad[4];

  const int t = threadIdx.x;
  const int g = blockIdx.x;       // b*N + i
  const int b = g >> 6;
  const int i = g & 63;

  for (int k = t; k < 896; k += 64) {
    sWh1e[k] = Wh1e[k];
    sWa1e[k] = Wa1e[k];
    sWv[k] = Wv[k];
  }
  for (int k = t; k < 256; k += 64) sWcomb[k] = Wcomb[k];
  for (int k = t; k < 128; k += 64) {
    sWa2[k] = Wa2[k];
    sBv[k] = bv[k];
    sPh[k] = Ph[(size_t)g * 128 + k];
    sPa[k] = Pa[(size_t)g * 128 + k];
    // copy ae into first half of decoder input row
    concat_h[(size_t)g * 256 + k] = ae_h[(size_t)g * 128 + k];
  }
  __syncthreads();

  float mask = 0.f;
  if (t < NE) {
    int j = t;
    int jj = j + (j >= i ? 1 : 0);
    const float* ei = emb + (size_t)g * 9;
    const float* ej = emb + (size_t)(b * 64 + jj) * 9;
    float dx = ej[0] - ei[0];
    float dy = ej[1] - ei[1];
    float sq = dx * dx + dy * dy;
    float dist = sq > 0.f ? sqrtf(sq) : 0.f;
    float ang = atan2f(dy, dx) - atan2f(ei[3], ei[2]);
    const float PI = 3.14159265358979323846f;
    float a = fmodf(ang + PI, 2.f * PI);
    if (a < 0.f) a += 2.f * PI;
    ang = a - PI;
    float edge[7];
    edge[0] = dist;
    edge[1] = cosf(ang);
    edge[2] = sinf(ang);
    edge[3] = ej[2];
    edge[4] = ej[3];
    edge[5] = ej[7];
    edge[6] = ej[8];
#pragma unroll
    for (int f = 0; f < 7; ++f) sEdge[j * 7 + f] = edge[f];

    out[OUT_URD + (size_t)g * NE + j] = 12.f * dist;

    // gate logits (Wh2/Whe folded into Wcomb) and attention score
    float hl0 = bcomb[0], hl1 = bcomb[1], sc = ba2[0];
    for (int d = 0; d < 128; ++d) {
      float eh = 0.f, ea = 0.f;
#pragma unroll
      for (int f = 0; f < 7; ++f) {
        eh += edge[f] * sWh1e[d * 7 + f];
        ea += edge[f] * sWa1e[d * 7 + f];
      }
      float h = fmaxf(sPh[d] + eh, 0.f);
      hl0 += h * sWcomb[d];
      hl1 += h * sWcomb[128 + d];
      float av = fmaxf(sPa[d] + ea, 0.f);
      sc += av * sWa2[d];
    }
    size_t gi = (size_t)g * NE + j;
    float g0 = gum[gi * 2 + 0];
    float g1 = gum[gi * 2 + 1];
    // y1 = softmax over 2 at temperature 0.5 -> sigmoid
    float y1 = 1.f / (1.f + expf(2.f * ((hl0 + g0) - (hl1 + g1))));
    out[OUT_HL + gi] = hl1;
    out[OUT_HW + gi] = y1;
    mask = y1 > 0.5f ? 1.f : 0.f;
    sMask[j] = mask;
    sMS[j] = mask > 0.f ? sc : -1e9f;
  }
  __syncthreads();

  if (t == 0) {
    float mx = -3.0e38f, msum = 0.f;
    for (int j = 0; j < NE; ++j) {
      mx = fmaxf(mx, sMS[j]);
      msum += sMask[j];
    }
    sBroad[0] = mx;
    sBroad[1] = msum;
  }
  __syncthreads();
  if (t < NE) sVal[t] = expf(sMS[t] - sBroad[0]);
  __syncthreads();
  if (t == 0) {
    float den = 0.f;
    for (int j = 0; j < NE; ++j) den += sVal[j];
    sBroad[2] = den;
  }
  __syncthreads();
  float combined = 0.f;
  if (t < NE) {
    bool all_zero = sBroad[1] < 1e-6f;
    float soft = all_zero ? 0.f : sVal[t] / sBroad[2];
    combined = soft * mask;
    sVal[t] = combined;
    out[OUT_CW + (size_t)g * NE + t] = combined;
  }
  __syncthreads();
  if (t == 0) {
    float sc2 = 0.f;
    for (int j = 0; j < NE; ++j) sc2 += sVal[j];
    sBroad[3] = sc2;
  }
  __syncthreads();
  if (t < NE) {
    float cwn = combined / (sBroad[3] + 1e-6f);
    sMS[t] = -cwn * logf(cwn + 1e-6f);
  }
  __syncthreads();
  if (t == 0) {
    float ent = 0.f;
    for (int j = 0; j < NE; ++j) ent += sMS[j];
    ent_ws[g] = ent;
    float w[7] = {0.f, 0.f, 0.f, 0.f, 0.f, 0.f, 0.f};
    float cs = 0.f;
    for (int j = 0; j < NE; ++j) {
      float cj = sVal[j];
      cs += cj;
#pragma unroll
      for (int f = 0; f < 7; ++f) w[f] += cj * sEdge[j * 7 + f];
    }
#pragma unroll
    for (int f = 0; f < 7; ++f) sMask[f] = w[f];
    sMask[7] = cs;
  }
  __syncthreads();
  // attn_out = (sum_j c_j * edge_j) @ Wv^T + (sum_j c_j) * bv  -> concat[:,128:256]
  for (int d = t; d < 128; d += 64) {
    float ao = sMask[7] * sBv[d];
#pragma unroll
    for (int f = 0; f < 7; ++f) ao += sMask[f] * sWv[d * 7 + f];
    concat_h[(size_t)g * 256 + 128 + d] = (f16)ao;
  }
}

// Deterministic entropy reduction over 8192 per-agent partials.
__global__ __launch_bounds__(256) void ent_reduce(const float* ent_ws, float* out) {
  __shared__ float s[256];
  float acc = 0.f;
  for (int i = threadIdx.x; i < MROW; i += 256) acc += ent_ws[i];
  s[threadIdx.x] = acc;
  __syncthreads();
  for (int st = 128; st > 0; st >>= 1) {
    if (threadIdx.x < st) s[threadIdx.x] += s[threadIdx.x + st];
    __syncthreads();
  }
  if (threadIdx.x == 0) out[OUT_ENT] = s[0] / (float)MROW;
}

// ---------------------------------------------------------------------------
extern "C" void kernel_launch(void* const* d_in, const int* in_sizes, int n_in,
                              void* d_out, int out_size, void* d_ws, size_t ws_size,
                              hipStream_t stream) {
  const float* emb = (const float*)d_in[0];
  const float* gum = (const float*)d_in[1];
  const float* W1  = (const float*)d_in[2];
  const float* b1  = (const float*)d_in[3];
  const float* W2  = (const float*)d_in[4];
  const float* b2  = (const float*)d_in[5];
  const float* Wh1 = (const float*)d_in[6];
  const float* bh1 = (const float*)d_in[7];
  const float* Wh2 = (const float*)d_in[8];
  const float* bh2 = (const float*)d_in[9];
  const float* Whe = (const float*)d_in[10];
  const float* bhe = (const float*)d_in[11];
  const float* Wq  = (const float*)d_in[12];
  const float* Wa1 = (const float*)d_in[13];
  const float* ba1 = (const float*)d_in[14];
  const float* Wa2 = (const float*)d_in[15];
  const float* ba2 = (const float*)d_in[16];
  const float* Wv  = (const float*)d_in[17];
  const float* bv  = (const float*)d_in[18];
  const float* Wd1 = (const float*)d_in[19];
  const float* bd1 = (const float*)d_in[20];
  const float* Wd2 = (const float*)d_in[21];
  const float* bd2 = (const float*)d_in[22];
  float* out = (float*)d_out;

  char* ws = (char*)d_ws;
  size_t off = 0;
  auto take = [&](size_t bytes) -> char* {
    char* p = ws + off;
    off = (off + bytes + 255) & ~(size_t)255;
    return p;
  };

  f16* h1h     = (f16*)take((size_t)MROW * 128 * 2);
  f16* ae_h    = (f16*)take((size_t)MROW * 128 * 2);
  f16* q_h     = (f16*)take((size_t)MROW * 128 * 2);
  f16* concat_h= (f16*)take((size_t)MROW * 256 * 2);
  f16* d1_h    = (f16*)take((size_t)MROW * 256 * 2);
  float* Ph    = (float*)take((size_t)MROW * 128 * 4);
  float* Pa    = (float*)take((size_t)MROW * 128 * 4);
  f16* W2h     = (f16*)take(128 * 128 * 2);
  f16* Wqh     = (f16*)take(128 * 128 * 2);
  f16* Wh1ah   = (f16*)take(128 * 128 * 2);
  f16* Wa1ah   = (f16*)take(128 * 128 * 2);
  f16* Wd1h    = (f16*)take(256 * 256 * 2);
  f16* Wd2h    = (f16*)take(256 * 256 * 2);
  float* Wh1e  = (float*)take(128 * 7 * 4);
  float* Wa1e  = (float*)take(128 * 7 * 4);
  float* Wcomb = (float*)take(2 * 128 * 4);
  float* bcomb = (float*)take(2 * 4);
  float* ent_ws= (float*)take((size_t)MROW * 4);

  prep_weights<<<64, 256, 0, stream>>>(W2, Wq, Wh1, Wa1, Wh2, bh2, Whe, bhe, Wd1, Wd2,
                                       W2h, Wqh, Wh1ah, Wh1e, Wa1ah, Wa1e, Wd1h, Wd2h,
                                       Wcomb, bcomb);
  h1_kernel<<<1024, 256, 0, stream>>>(emb, W1, b1, h1h);

  // ae = lrelu(h1 @ W2^T + b2)          (M=8192, K=128, N=128)
  wmma_gemm<<<1024, 128, 0, stream>>>(h1h, W2h, b2, nullptr, ae_h, 128, 128, 8, 1);
  // q = ae @ Wq^T
  wmma_gemm<<<1024, 128, 0, stream>>>(ae_h, Wqh, nullptr, nullptr, q_h, 128, 128, 8, 0);
  // P_h = ae @ Wh1[:, :128]^T + bh1   (relu deferred until edge term added)
  wmma_gemm<<<1024, 128, 0, stream>>>(ae_h, Wh1ah, bh1, Ph, nullptr, 128, 128, 8, 0);
  // P_a = q @ Wa1[:, :128]^T + ba1
  wmma_gemm<<<1024, 128, 0, stream>>>(q_h, Wa1ah, ba1, Pa, nullptr, 128, 128, 8, 0);

  edge_kernel<<<MROW, 64, 0, stream>>>(emb, gum, Ph, Pa, Wh1e, Wa1e, Wcomb, bcomb,
                                       Wa2, ba2, Wv, bv, ae_h, concat_h, out, ent_ws);

  // d1 = lrelu(concat @ Wd1^T + bd1)    (M=8192, K=256, N=256)
  wmma_gemm<<<2048, 128, 0, stream>>>(concat_h, Wd1h, bd1, nullptr, d1_h, 256, 256, 16, 1);
  // att = lrelu(d1 @ Wd2^T + bd2) -> d_out[OUT_ATT]
  wmma_gemm<<<2048, 128, 0, stream>>>(d1_h, Wd2h, bd2, out + OUT_ATT, nullptr, 256, 256, 16, 1);

  ent_reduce<<<1, 256, 0, stream>>>(ent_ws, out);
}
